// RIM_19542101197255
// MI455X (gfx1250) — compile-verified, matrix operand order checked by reading, not run on hardware
//
#include <hip/hip_runtime.h>
#include <hip/hip_bf16.h>

// ---------------------------------------------------------------------------
// RIMs block for MI455X (gfx1250): bf16 WMMA GEMM cores, fp32 accumulate.
//   S=512, B=32, R=8, H=128, TOP=3, CHL=19, I=20
// ---------------------------------------------------------------------------

#define S_LEN 512
#define B_SZ  32
#define R_SZ  8
#define H_SZ  128
#define CHL   19
#define I_SZ  20
#define G3H   384   // 3*H

typedef __attribute__((ext_vector_type(16))) __bf16 v16bf;
typedef __attribute__((ext_vector_type(8)))  float  v8f;

__device__ __forceinline__ unsigned short f2bf(float f) {
    unsigned int u = __float_as_uint(f);
    u += 0x7FFFu + ((u >> 16) & 1u);          // round-to-nearest-even
    return (unsigned short)(u >> 16);
}

#if __has_builtin(__builtin_amdgcn_cvt_pk_bf16_f32)
__device__ __forceinline__ unsigned int pack_bf2(float lo, float hi) {
    auto r = __builtin_amdgcn_cvt_pk_bf16_f32(lo, hi);   // one VALU op
    unsigned int u;
    __builtin_memcpy(&u, &r, sizeof(u));
    return u;
}
#else
__device__ __forceinline__ unsigned int pack_bf2(float lo, float hi) {
    return (unsigned int)f2bf(lo) | ((unsigned int)f2bf(hi) << 16);
}
#endif

// Fast nonlinearities: v_exp_f32 + v_rcp_f32 only (no IEEE div expansion).
__device__ __forceinline__ float fast_sigmoid(float x) {
    return __builtin_amdgcn_rcpf(1.f + __expf(-x));
}
__device__ __forceinline__ float fast_tanh(float x) {
    // tanh(x) = 1 - 2/(e^{2x}+1); saturates correctly at +-inf
    return 1.f - 2.f * __builtin_amdgcn_rcpf(1.f + __expf(2.f * x));
}

// Per-row fragment base: one address register per row, all loads at constant
// dword offsets (DS immediate range), so no per-load address VALU.
__device__ __forceinline__ const unsigned int* frag_row(const unsigned short* base,
                                                        int row, int stride, int lane) {
    return (const unsigned int*)(base + row * stride + ((lane >> 4) << 3));
}
// 16x32 bf16 WMMA fragment, K-chunk kc, from a per-row base pointer.
// ISA layout (CDNA5 ISA): lanes 0-15: K = kbase+{0..7,16..23};
// lanes 16-31: K = kbase+{8..15,24..31}; pairs contiguous.
__device__ __forceinline__ v16bf load_frag_idx(const unsigned int* rowp, int kc) {
    union { unsigned int u[8]; v16bf v; } f;
#pragma unroll
    for (int j = 0; j < 4; ++j) f.u[j]     = rowp[kc * 16 + j];
#pragma unroll
    for (int j = 0; j < 4; ++j) f.u[4 + j] = rowp[kc * 16 + 8 + j];
    return f.v;
}

// Fragment built from a row-major fp32 array (global weights), converting to
// bf16 on the fly.  Used once per block to make weights register-resident.
__device__ __forceinline__ v16bf load_frag_f32(const float* base, int row,
                                               int stride, int kbase, int lane) {
    union { unsigned int u[8]; v16bf v; } f;
    const float* p = base + (size_t)row * stride + kbase + ((lane >> 4) << 3);
#pragma unroll
    for (int j = 0; j < 4; ++j) f.u[j]     = pack_bf2(p[2 * j],      p[2 * j + 1]);
#pragma unroll
    for (int j = 0; j < 4; ++j) f.u[4 + j] = pack_bf2(p[16 + 2 * j], p[16 + 2 * j + 1]);
    return f.v;
}

// ---------------------------------------------------------------------------
// Kernel 1: k/v input projections.  k[b,i,o] = sum_s x[b,i-1,s]*Kir_w[o,s]
// (i==0 is the null channel -> 0).  Pure HBM roofline.
// ---------------------------------------------------------------------------
__global__ void __launch_bounds__(256) kv_kernel(const float* __restrict__ x,
                                                 const float* __restrict__ Kw,
                                                 const float* __restrict__ Vw,
                                                 float* __restrict__ kbuf,
                                                 float* __restrict__ vbuf) {
    const int total = 2 * B_SZ * I_SZ * H_SZ;
    int gid = blockIdx.x * blockDim.x + threadIdx.x;
    if (gid >= total) return;
    int which = gid / (B_SZ * I_SZ * H_SZ);
    int rem   = gid % (B_SZ * I_SZ * H_SZ);
    int b = rem / (I_SZ * H_SZ);
    int i = (rem / H_SZ) % I_SZ;
    int o = rem % H_SZ;
    float acc = 0.f;
    if (i > 0) {
        const float* xr = x + ((size_t)(b * CHL + (i - 1))) * S_LEN;
        const float* wr = (which ? Vw : Kw) + (size_t)o * S_LEN;
#pragma unroll 4
        for (int s = 0; s < S_LEN; s += 4) {
            float4 xa = *(const float4*)(xr + s);
            float4 wa = *(const float4*)(wr + s);
            acc += xa.x * wa.x + xa.y * wa.y + xa.z * wa.z + xa.w * wa.w;
        }
    }
    (which ? vbuf : kbuf)[(b * I_SZ + i) * H_SZ + o] = acc;
}

// ---------------------------------------------------------------------------
// Kernel 2: per-rim GRU scan.  One persistent block per rim, 8 waves.
// Wave w owns output columns hh in [16w, 16w+16) for ALL three gates
// (nj tiles {w, w+8, w+16}): 15 B-fragments stay in VGPRs for the whole scan,
// gates run directly on WMMA accumulators, hidden state in 16 regs/lane.
// LDS: Wih padded (24KB) + xpad (2KB) + h_bf (8KB) = 34KB.
// ---------------------------------------------------------------------------
__global__ void __launch_bounds__(256) gru_kernel(const float* __restrict__ x,
                                                  const float* __restrict__ Wih,
                                                  const float* __restrict__ Whh,
                                                  float* __restrict__ h_out) {
    extern __shared__ unsigned char smem[];
    unsigned short* Wih_bf = (unsigned short*)smem;            // 384*32 (K pad)
    unsigned short* xpad   = Wih_bf + G3H * 32;                // 32*32
    unsigned short* h_bf   = xpad + 32 * 32;                   // 32*128

    const int tid  = threadIdx.x;
    const int lane = tid & 31;
    const int wid  = tid >> 5;
    const int rim  = blockIdx.x;

    // stage Wih zero-padded to K=32 (pair stores; I_SZ=20 is even)
    for (int pi = tid; pi < G3H * 16; pi += 256) {
        int g = pi >> 4, ip = (pi & 15) * 2;
        unsigned int u = 0u;
        if (ip < I_SZ) {
            const float* w = Wih + ((size_t)rim * G3H + g) * I_SZ + ip;
            u = pack_bf2(w[0], w[1]);
        }
        *(unsigned int*)(Wih_bf + g * 32 + ip) = u;
    }
    for (int idx = tid; idx < B_SZ * H_SZ / 2; idx += 256)
        ((unsigned int*)h_bf)[idx] = 0u;
    __syncthreads();

    // register-resident B fragments: gate g, K-chunk kc
    const float* WhhR = Whh + (size_t)rim * G3H * H_SZ;
    v16bf Bh[3][4];
    v16bf Bx[3];
#pragma unroll
    for (int g = 0; g < 3; ++g) {
        int brow = g * H_SZ + wid * 16 + (lane & 15);
#pragma unroll
        for (int kc = 0; kc < 4; ++kc)
            Bh[g][kc] = load_frag_f32(WhhR, brow, H_SZ, kc * 32, lane);
        Bx[g] = load_frag_idx(frag_row(Wih_bf, brow, 32, lane), 0);
    }

    // per-lane hidden state (C layout: rows rbase..rbase+7, col)
    float hprev[2][8];
#pragma unroll
    for (int mi = 0; mi < 2; ++mi)
#pragma unroll
        for (int v = 0; v < 8; ++v) hprev[mi][v] = 0.f;

    const int col = wid * 16 + (lane & 15);
    unsigned short* hst = h_bf + col;                         // store base (LDS)
    float* ost = h_out + (size_t)rim * H_SZ + col;            // store base (global)

    for (int s = 0; s < S_LEN; ++s) {
        // stage x_t -> padded bf16, pair stores (null ch i==0, pad i>=20 zero)
        for (int pi = tid; pi < 32 * 16; pi += 256) {
            int b = pi >> 4, ip = (pi & 15) * 2;
            float v0 = (ip >= 1 && ip < I_SZ)
                         ? x[((size_t)(b * CHL + ip - 1)) * S_LEN + s] : 0.f;
            float v1 = (ip + 1 < I_SZ)
                         ? x[((size_t)(b * CHL + ip)) * S_LEN + s] : 0.f;
            *(unsigned int*)(xpad + b * 32 + ip) = pack_bf2(v0, v1);
        }
        __syncthreads();   // xpad ready; previous-step h_bf stores visible

        // A fragments (shared across the wave's 3 gate tiles)
        v16bf Ah[2][4], Ax[2];
#pragma unroll
        for (int mi = 0; mi < 2; ++mi) {
            int arow = mi * 16 + (lane & 15);
            const unsigned int* hp = frag_row(h_bf, arow, H_SZ, lane);
#pragma unroll
            for (int kc = 0; kc < 4; ++kc)
                Ah[mi][kc] = load_frag_idx(hp, kc);
            Ax[mi] = load_frag_idx(frag_row(xpad, arow, 32, lane), 0);
        }

#pragma unroll
        for (int mi = 0; mi < 2; ++mi) {
            v8f aR = {}, aZ = {}, aN = {};
#pragma unroll
            for (int kc = 0; kc < 4; ++kc) {
                aR = __builtin_amdgcn_wmma_f32_16x16x32_bf16(
                         false, Ah[mi][kc], false, Bh[0][kc], (short)0, aR, false, false);
                aZ = __builtin_amdgcn_wmma_f32_16x16x32_bf16(
                         false, Ah[mi][kc], false, Bh[1][kc], (short)0, aZ, false, false);
                aN = __builtin_amdgcn_wmma_f32_16x16x32_bf16(
                         false, Ah[mi][kc], false, Bh[2][kc], (short)0, aN, false, false);
            }
            v8f xR = {}, xZ = {}, xN = {};
            xR = __builtin_amdgcn_wmma_f32_16x16x32_bf16(
                     false, Ax[mi], false, Bx[0], (short)0, xR, false, false);
            xZ = __builtin_amdgcn_wmma_f32_16x16x32_bf16(
                     false, Ax[mi], false, Bx[1], (short)0, xZ, false, false);
            xN = __builtin_amdgcn_wmma_f32_16x16x32_bf16(
                     false, Ax[mi], false, Bx[2], (short)0, xN, false, false);
            // gates (torch order r,z,n): h = (1-z)*tanh(xn + r*hn) + z*h
#pragma unroll
            for (int v = 0; v < 8; ++v) {
                float r = fast_sigmoid(xR[v] + aR[v]);
                float z = fast_sigmoid(xZ[v] + aZ[v]);
                float n = fast_tanh(xN[v] + r * aN[v]);
                hprev[mi][v] = (1.f - z) * n + z * hprev[mi][v];
            }
        }
        __syncthreads();   // all waves done reading h_bf of step s

        // write new hidden state: bf16 copy for next step's A frags + global.
        // single base + constant offsets per store.
        float* osts = ost + (size_t)s * B_SZ * R_SZ * H_SZ;
#pragma unroll
        for (int mi = 0; mi < 2; ++mi) {
            int rbase = mi * 16 + ((lane >> 4) << 3);
#pragma unroll
            for (int v = 0; v < 8; ++v) {
                float hv = hprev[mi][v];
                hst[(rbase + v) * H_SZ] = f2bf(hv);
                osts[(size_t)(rbase + v) * R_SZ * H_SZ] = hv;
            }
        }
        // top-of-loop __syncthreads() makes these stores visible before reads
    }
}

// ---------------------------------------------------------------------------
// Kernel 3: attention stages, 2 tokens per block (M = 16 = 2 tokens x 8 rims).
// ---------------------------------------------------------------------------
__device__ __forceinline__ void gemm_16x128x128(const v16bf A[4],
                                                const unsigned short* W,
                                                float* Cf, const float* bias,
                                                unsigned short* Cbf,
                                                int lane, int wid) {
    v8f acc = {};
    const unsigned int* wrow =
        frag_row(W, wid * 16 + (lane & 15), H_SZ, lane);
#pragma unroll
    for (int kc = 0; kc < 4; ++kc) {
        v16bf b = load_frag_idx(wrow, kc);
        acc = __builtin_amdgcn_wmma_f32_16x16x32_bf16(
                  false, A[kc], false, b, (short)0, acc, false, false);
    }
    int col = wid * 16 + (lane & 15);
    int rb  = (lane >> 4) << 3;
#pragma unroll
    for (int v = 0; v < 8; ++v) {
        float val = acc[v];
        if (bias) val += bias[col];
        if (Cf)  Cf[(rb + v) * H_SZ + col] = val;
        if (Cbf) Cbf[(rb + v) * H_SZ + col] = f2bf(val);
    }
}

__global__ void __launch_bounds__(256) attn_kernel(const float* __restrict__ hbuf,
                                                   const float* __restrict__ kbuf,
                                                   const float* __restrict__ vbuf,
                                                   const float* __restrict__ Qir_w,
                                                   const float* __restrict__ Qir_b,
                                                   const float* __restrict__ Qrr_w,
                                                   const float* __restrict__ Krr_w,
                                                   const float* __restrict__ Vrr_w,
                                                   float* __restrict__ out) {
    extern __shared__ unsigned char smem[];
    unsigned short* Wq   = (unsigned short*)smem;          // 128*128 each, bf16
    unsigned short* Wq2  = Wq  + H_SZ * H_SZ;
    unsigned short* Wk2  = Wq2 + H_SZ * H_SZ;
    unsigned short* Wv2  = Wk2 + H_SZ * H_SZ;
    unsigned short* hrow = Wv2 + H_SZ * H_SZ;              // 16*128
    unsigned short* h1bf = hrow + 16 * H_SZ;
    unsigned short* q2bf = h1bf + 16 * H_SZ;
    unsigned short* k2bf = q2bf + 16 * H_SZ;
    float* q_f    = (float*)(k2bf + 16 * H_SZ);            // 16*128
    float* h1_f   = q_f  + 16 * H_SZ;
    float* v2_f   = h1_f + 16 * H_SZ;
    float* qk_l   = v2_f + 16 * H_SZ;                      // 2*8*20 (logits->soft)
    float* mask_l = qk_l + 2 * R_SZ * I_SZ;                // 16
    float* qk2_f  = mask_l + 16;                           // 16*16
    float* sf2_l  = qk2_f + 256;                           // 2*8*8
    float* bias_l = sf2_l + 2 * R_SZ * R_SZ;               // 128

    const int tid  = threadIdx.x;
    const int lane = tid & 31;
    const int wid  = tid >> 5;
    const int t0   = blockIdx.x * 2;                       // first token (s*32+b)

    // weight staging: packed-pair converts + b32 stores
    for (int pi = tid; pi < H_SZ * H_SZ / 2; pi += 256) {
        int idx = pi * 2;
        *(unsigned int*)(Wq  + idx) = pack_bf2(Qir_w[idx], Qir_w[idx + 1]);
        *(unsigned int*)(Wq2 + idx) = pack_bf2(Qrr_w[idx], Qrr_w[idx + 1]);
        *(unsigned int*)(Wk2 + idx) = pack_bf2(Krr_w[idx], Krr_w[idx + 1]);
        *(unsigned int*)(Wv2 + idx) = pack_bf2(Vrr_w[idx], Vrr_w[idx + 1]);
    }
    if (tid < H_SZ) bias_l[tid] = Qir_b[tid];
    for (int pi = tid; pi < 16 * H_SZ / 2; pi += 256) {
        int idx = pi * 2;
        int m = idx >> 7, hh = idx & 127;                  // m = tok*8 + r
        int T = t0 + (m >> 3);
        const float* src = hbuf + (((size_t)T) * R_SZ + (m & 7)) * H_SZ + hh;
        *(unsigned int*)(hrow + idx) = pack_bf2(src[0], src[1]);
    }
    __syncthreads();

    // q = h @ Qir^T + b
    {
        const unsigned int* hp = frag_row(hrow, lane & 15, H_SZ, lane);
        v16bf Afr[4];
#pragma unroll
        for (int kc = 0; kc < 4; ++kc) Afr[kc] = load_frag_idx(hp, kc);
        gemm_16x128x128(Afr, Wq, q_f, bias_l, nullptr, lane, wid);
    }
    __syncthreads();

    // qk[tok,r,i] = q . k / sqrt(H)   (i==0 hits the null channel -> 0)
    for (int idx = tid; idx < 2 * R_SZ * I_SZ; idx += 256) {
        int tok = idx / (R_SZ * I_SZ);
        int r   = (idx / I_SZ) & 7;
        int i   = idx % I_SZ;
        float d = 0.f;
        if (i > 0) {
            int b = (t0 + tok) & 31;
            const float* qp = q_f + (tok * 8 + r) * H_SZ;
            const float* kp = kbuf + (size_t)(b * I_SZ + i) * H_SZ;
#pragma unroll 4
            for (int h = 0; h < H_SZ; ++h) d += qp[h] * kp[h];
        }
        qk_l[idx] = d * 0.088388347648318447f;             // 1/sqrt(128)
    }
    __syncthreads();

    // softmax over i (in place)
    if (tid < 16) {
        float* row = qk_l + (tid >> 3) * R_SZ * I_SZ + (tid & 7) * I_SZ;
        float mx = row[0];
        for (int i = 1; i < I_SZ; ++i) mx = fmaxf(mx, row[i]);
        float sum = 0.f;
        for (int i = 0; i < I_SZ; ++i) { row[i] = __expf(row[i] - mx); sum += row[i]; }
        float inv = __builtin_amdgcn_rcpf(sum);
        for (int i = 0; i < I_SZ; ++i) row[i] *= inv;
    }
    __syncthreads();

    // top-3 smallest soft[:,0] per token -> mask (rank count == top_k tiebreak)
    if (tid < 16) {
        int tok = tid >> 3, r = tid & 7;
        const float* s0 = qk_l + tok * R_SZ * I_SZ;
        float mine = s0[r * I_SZ];
        int rank = 0;
        for (int r2 = 0; r2 < R_SZ; ++r2) {
            float o = s0[r2 * I_SZ];
            rank += (o < mine) || (o == mine && r2 < r);
        }
        mask_l[tid] = (rank < 3) ? 1.f : 0.f;
    }
    __syncthreads();

    // h1 = soft @ v  (forward path of the NoGrad trick is identity)
    for (int pi = tid; pi < 16 * H_SZ / 2; pi += 256) {
        int idx = pi * 2;
        int m = idx >> 7, o = idx & 127;
        int tok = m >> 3, r = m & 7;
        int b = (t0 + tok) & 31;
        const float* sp = qk_l + tok * R_SZ * I_SZ + r * I_SZ;
        const float* vp = vbuf + (size_t)(b * I_SZ) * H_SZ + o;
        float a0 = 0.f, a1 = 0.f;
        for (int i = 1; i < I_SZ; ++i) {
            float sv = sp[i];
            a0 += sv * vp[i * H_SZ];
            a1 += sv * vp[i * H_SZ + 1];
        }
        h1_f[idx]     = a0;
        h1_f[idx + 1] = a1;
        *(unsigned int*)(h1bf + idx) = pack_bf2(a0, a1);
    }
    __syncthreads();

    // stage two projections (shared A fragments)
    {
        const unsigned int* hp = frag_row(h1bf, lane & 15, H_SZ, lane);
        v16bf A1[4];
#pragma unroll
        for (int kc = 0; kc < 4; ++kc) A1[kc] = load_frag_idx(hp, kc);
        gemm_16x128x128(A1, Wq2, nullptr, nullptr, q2bf, lane, wid);
        gemm_16x128x128(A1, Wk2, nullptr, nullptr, k2bf, lane, wid);
        gemm_16x128x128(A1, Wv2, v2_f,   nullptr, nullptr, lane, wid);
    }
    __syncthreads();

    // qk2 = q2 @ k2^T : one 16x16 tile; only the two diagonal 8x8 blocks
    // (per-token) are meaningful.
    if (wid == 0) {
        const unsigned int* qp = frag_row(q2bf, lane & 15, H_SZ, lane);
        const unsigned int* kp = frag_row(k2bf, lane & 15, H_SZ, lane);
        v8f acc = {};
#pragma unroll
        for (int kc = 0; kc < 4; ++kc) {
            acc = __builtin_amdgcn_wmma_f32_16x16x32_bf16(
                      false, load_frag_idx(qp, kc), false, load_frag_idx(kp, kc),
                      (short)0, acc, false, false);
        }
        int cl = lane & 15, rb = (lane >> 4) << 3;
#pragma unroll
        for (int v = 0; v < 8; ++v) qk2_f[(rb + v) * 16 + cl] = acc[v];
    }
    __syncthreads();

    // soft2 = softmax_c(qk2) * mask[r]
    if (tid < 16) {
        int tok = tid >> 3, r = tid & 7;
        const float* row = qk2_f + (tok * 8 + r) * 16 + tok * 8;
        float mx = row[0];
        for (int c = 1; c < R_SZ; ++c) mx = fmaxf(mx, row[c]);
        float e[R_SZ], sum = 0.f;
        for (int c = 0; c < R_SZ; ++c) { e[c] = __expf(row[c] - mx); sum += e[c]; }
        float sc = mask_l[tid] * __builtin_amdgcn_rcpf(sum);
        for (int c = 0; c < R_SZ; ++c) sf2_l[tid * R_SZ + c] = e[c] * sc;
    }
    __syncthreads();

    // out = (h1 + soft2 @ v2) * mask
    for (int idx = tid; idx < 16 * H_SZ; idx += 256) {
        int m = idx >> 7, o = idx & 127;
        int tok = m >> 3, r = m & 7;
        const float* sp = sf2_l + m * R_SZ;
        float acc = h1_f[idx];
        for (int c = 0; c < R_SZ; ++c) acc += sp[c] * v2_f[(tok * 8 + c) * H_SZ + o];
        acc *= mask_l[m];
        out[(((size_t)(t0 + tok)) * R_SZ + r) * H_SZ + o] = acc;
    }
}

// ---------------------------------------------------------------------------
extern "C" void kernel_launch(void* const* d_in, const int* in_sizes, int n_in,
                              void* d_out, int out_size, void* d_ws, size_t ws_size,
                              hipStream_t stream) {
    const float* x     = (const float*)d_in[0];
    const float* Wih   = (const float*)d_in[1];
    const float* Whh   = (const float*)d_in[2];
    const float* Qir_w = (const float*)d_in[3];
    const float* Qir_b = (const float*)d_in[4];
    const float* Kir_w = (const float*)d_in[5];
    const float* Vir_w = (const float*)d_in[6];
    const float* Qrr_w = (const float*)d_in[7];
    const float* Krr_w = (const float*)d_in[8];
    const float* Vrr_w = (const float*)d_in[9];
    float* out = (float*)d_out;

    // workspace: h (S,B,R,H) fp32 + k,v (B,I,H) fp32  (~67.8 MB)
    float* h_buf = (float*)d_ws;
    float* kbuf  = h_buf + (size_t)S_LEN * B_SZ * R_SZ * H_SZ;
    float* vbuf  = kbuf + (size_t)B_SZ * I_SZ * H_SZ;

    // dynamic LDS sizes (bytes)
    const size_t GRU_LDS  = (size_t)(G3H * 32 + 32 * 32 + B_SZ * H_SZ) * 2;  // 34816
    const size_t ATTN_LDS = (size_t)(4 * H_SZ * H_SZ + 4 * 16 * H_SZ) * 2
                          + (size_t)(3 * 16 * H_SZ + 2 * R_SZ * I_SZ + 16
                                     + 256 + 2 * R_SZ * R_SZ + H_SZ) * 4;    // 175424

    kv_kernel<<<(2 * B_SZ * I_SZ * H_SZ + 255) / 256, 256, 0, stream>>>(
        x, Kir_w, Vir_w, kbuf, vbuf);
    gru_kernel<<<R_SZ, 256, GRU_LDS, stream>>>(x, Wih, Whh, h_buf);
    attn_kernel<<<(S_LEN * B_SZ) / 2, 256, ATTN_LDS, stream>>>(
        h_buf, kbuf, vbuf, Qir_w, Qir_b, Qrr_w, Krr_w, Vrr_w, out);
}
